// AdaptiveInput_43550968381892
// MI455X (gfx1250) — compile-verified, matrix operand order checked by reading, not run on hardware
//
#include <hip/hip_runtime.h>

#define D_OUT 1024
#define CUT1 10000
#define CUT2 60000
#define CUT3 190000

typedef __attribute__((ext_vector_type(16))) __bf16        v16bf;
typedef __attribute__((ext_vector_type(8)))  float         v8f;
typedef __attribute__((ext_vector_type(4)))  float         f4;
typedef __attribute__((ext_vector_type(2)))  unsigned int  u32x2;
typedef __attribute__((ext_vector_type(4)))  unsigned int  u32x4;

__device__ __forceinline__ unsigned short cvt_bf16(float f) {
    union { float f; unsigned int u; } v; v.f = f;
    unsigned int u = v.u + (0x7FFFu + ((v.u >> 16) & 1u));   // RNE
    return (unsigned short)(u >> 16);
}
__device__ __forceinline__ unsigned int pack2_bf16(float lo, float hi) {
    union { float f; unsigned int u; } a, b; a.f = lo; b.f = hi;
    unsigned int ua = a.u + (0x7FFFu + ((a.u >> 16) & 1u));
    unsigned int ub = b.u + (0x7FFFu + ((b.u >> 16) & 1u));
    return (ua >> 16) | (ub & 0xFFFF0000u);
}

__global__ void ai_init_counters(int* counters) {
    if (threadIdx.x < 4) counters[threadIdx.x] = 0;
}

__global__ void ai_partition(const int* __restrict__ tokens, int n,
                             int* __restrict__ counters, int* __restrict__ lists,
                             int cap) {
    int i = blockIdx.x * blockDim.x + threadIdx.x;
    if (i >= n) return;
    int t = tokens[i];
    int c = (t < CUT1) ? 0 : (t < CUT2) ? 1 : (t < CUT3) ? 2 : 3;
    int pos = atomicAdd(&counters[c], 1);
    if (pos < cap) lists[c * cap + pos] = i;
}

// f32 [N][K] row-major -> bf16 [N][K] (head_w is already [d][h] = [n][k])
__global__ void ai_conv_straight(const float* __restrict__ src,
                                 unsigned short* __restrict__ dst, int total) {
    int g = (blockIdx.x * blockDim.x + threadIdx.x) * 8;
    if (g >= total) return;
    f4 a = *(const f4*)(src + g);
    f4 b = *(const f4*)(src + g + 4);
    u32x4 o;
    o.x = pack2_bf16(a.x, a.y); o.y = pack2_bf16(a.z, a.w);
    o.z = pack2_bf16(b.x, b.y); o.w = pack2_bf16(b.z, b.w);
    *(u32x4*)(dst + g) = o;
}

// f32 [K][1024] row-major (tail lin) -> bf16 [1024][K] (transposed to [n][k])
__global__ void ai_conv_transpose(const float* __restrict__ src,
                                  unsigned short* __restrict__ dst, int K) {
    int g = (blockIdx.x * blockDim.x + threadIdx.x) * 8;
    if (g >= K * D_OUT) return;
    int k = g >> 10, n0 = g & 1023;               // n fastest: coalesced reads
    f4 a = *(const f4*)(src + (long)k * D_OUT + n0);
    f4 b = *(const f4*)(src + (long)k * D_OUT + n0 + 4);
    dst[(long)(n0 + 0) * K + k] = cvt_bf16(a.x);
    dst[(long)(n0 + 1) * K + k] = cvt_bf16(a.y);
    dst[(long)(n0 + 2) * K + k] = cvt_bf16(a.z);
    dst[(long)(n0 + 3) * K + k] = cvt_bf16(a.w);
    dst[(long)(n0 + 4) * K + k] = cvt_bf16(b.x);
    dst[(long)(n0 + 5) * K + k] = cvt_bf16(b.y);
    dst[(long)(n0 + 6) * K + k] = cvt_bf16(b.z);
    dst[(long)(n0 + 7) * K + k] = cvt_bf16(b.w);
}

// ---------- main GEMM: preconverted bf16 weights, double-buffered LDS ----------
template<int HSZ>
__global__ __launch_bounds__(256)
void ai_cluster_gemm_pre(const int* __restrict__ tokens,
                         const float* __restrict__ emb,
                         const unsigned short* __restrict__ wbf,   // [1024][HSZ] bf16
                         const int* __restrict__ counters, int cluster,
                         const int* __restrict__ list, int cap,
                         int low, float* __restrict__ out)
{
    constexpr int KC = 32, NB = 128;
    constexpr int KITERS = (HSZ + KC - 1) / KC;
    constexpr int ASZ = 16 * KC, BSZ = NB * KC;

    __shared__ unsigned short ldsA[2 * ASZ];      // ping-pong
    __shared__ unsigned short ldsB[2 * BSZ];
    __shared__ float ldsC[16 * NB];
    __shared__ int s_pos[16], s_row[16], s_valid[16];

    int count = counters[cluster];
    if (count > cap) count = cap;
    int m0 = blockIdx.x * 16;
    if (m0 >= count) return;
    int n0 = blockIdx.y * NB;
    int tid = threadIdx.x;

    if (tid < 16) {
        int g = m0 + tid;
        int valid = (g < count) ? 1 : 0;
        int pos = valid ? list[g] : 0;
        s_pos[tid]   = pos;
        s_row[tid]   = valid ? (tokens[pos] - low) : 0;
        s_valid[tid] = valid;
    }
    __syncthreads();

    // A ownership: tid<128 -> row aM, 4 consecutive k at aK; hoist row base ptr
    const int aM = tid >> 3, aK = (tid & 7) << 2;
    const float* aPtr = emb + (long)s_row[aM] * HSZ + aK;   // s_row read ONCE

    // B ownership: 512 16B slots; thread owns slots tid, tid+256; hoist ptrs
    const int bN0 = tid >> 2,          bK0 = (tid & 3) << 3;
    const int bN1 = (tid + 256) >> 2,  bK1 = ((tid + 256) & 3) << 3;
    const unsigned short* bPtr0 = wbf + (long)(n0 + bN0) * HSZ + bK0;
    const unsigned short* bPtr1 = wbf + (long)(n0 + bN1) * HSZ + bK1;

    auto loadA = [&](int k0) -> f4 {
        f4 r = {0.f, 0.f, 0.f, 0.f};
        if (tid < 128 && (k0 + aK + 4 <= HSZ)) r = *(const f4*)(aPtr + k0);
        return r;
    };
    auto loadB0 = [&](int k0) -> u32x4 {
        u32x4 r = {0u, 0u, 0u, 0u};
        if (k0 + bK0 < HSZ) r = *(const u32x4*)(bPtr0 + k0);
        return r;
    };
    auto loadB1 = [&](int k0) -> u32x4 {
        u32x4 r = {0u, 0u, 0u, 0u};
        if (k0 + bK1 < HSZ) r = *(const u32x4*)(bPtr1 + k0);
        return r;
    };
    auto stage = [&](int buf, f4 ra, u32x4 rb0, u32x4 rb1) {
        if (tid < 128) {
            u32x2 t; t.x = pack2_bf16(ra.x, ra.y); t.y = pack2_bf16(ra.z, ra.w);
            *(u32x2*)&ldsA[buf * ASZ + aM * KC + aK] = t;
        }
        *(u32x4*)&ldsB[buf * BSZ + bN0 * KC + bK0] = rb0;   // pure b128 copies
        *(u32x4*)&ldsB[buf * BSZ + bN1 * KC + bK1] = rb1;
    };

    // prologue: stage chunk0 into buf0, prefetch chunk1
    f4 regA = loadA(0);
    u32x4 regB0 = loadB0(0), regB1 = loadB1(0);
    stage(0, regA, regB0, regB1);
    if (KITERS > 1) { regA = loadA(KC); regB0 = loadB0(KC); regB1 = loadB1(KC); }
    __syncthreads();

    const int lane = tid & 31, wave = tid >> 5;
    const int mn = lane & 15, hi = (lane >> 4) & 1;
    v8f acc = {};

    for (int kt = 0; kt < KITERS; ++kt) {
        int buf = kt & 1;
        // fragment reads first (LDS in-order per wave: keep reads ahead of stores)
        union { unsigned int u[8]; v16bf v; } fa, fb;
        const unsigned int* pa = (const unsigned int*)(ldsA + buf * ASZ + mn * KC);
        const unsigned int* pb = (const unsigned int*)(ldsB + buf * BSZ + (wave * 16 + mn) * KC);
        #pragma unroll
        for (int p = 0; p < 8; ++p) {
            int kb = 2 * p + ((p >= 4) ? 8 : 0) + hi * 8;
            fa.u[p] = pa[kb >> 1];
            fb.u[p] = pb[kb >> 1];
        }
        if (kt + 1 < KITERS) stage(buf ^ 1, regA, regB0, regB1);   // other buffer: safe
        if (kt + 2 < KITERS) {                                      // prefetch depth 2
            int k0 = (kt + 2) * KC;
            regA = loadA(k0); regB0 = loadB0(k0); regB1 = loadB1(k0);
        }
        acc = __builtin_amdgcn_wmma_f32_16x16x32_bf16(false, fa.v, false, fb.v,
                                                      (short)0, acc, false, false);
        if (kt + 1 < KITERS) __syncthreads();    // single barrier per K-step
    }

    __syncthreads();                              // protect ldsC reuse region
    #pragma unroll
    for (int r = 0; r < 8; ++r)
        ldsC[(r + hi * 8) * NB + wave * 16 + mn] = acc[r];
    __syncthreads();

    #pragma unroll
    for (int i = 0; i < 2; ++i) {
        int e = tid + i * 256;
        int m = e >> 5, cq = (e & 31) << 2;
        if (s_valid[m]) {
            f4 v = *(const f4*)&ldsC[m * NB + cq];
            __builtin_nontemporal_store(v, (f4*)(out + (long)s_pos[m] * D_OUT + n0 + cq));
        }
    }
}

// ---------- fallback: convert weights in-kernel (small ws) ----------
template<int HSZ, bool HEAD>
__global__ __launch_bounds__(256)
void ai_cluster_gemm_cvt(const int* __restrict__ tokens,
                         const float* __restrict__ emb,
                         const float* __restrict__ w,
                         const int* __restrict__ counters, int cluster,
                         const int* __restrict__ list, int cap,
                         int low, float* __restrict__ out)
{
    constexpr int KC = 32, NB = 128;
    constexpr int KITERS = (HSZ + KC - 1) / KC;

    __shared__ unsigned short ldsA[16 * KC];
    __shared__ unsigned short ldsB[NB * KC];
    __shared__ float ldsC[16 * NB];
    __shared__ int s_pos[16], s_row[16], s_valid[16];

    int count = counters[cluster];
    if (count > cap) count = cap;
    int m0 = blockIdx.x * 16;
    if (m0 >= count) return;
    int n0 = blockIdx.y * NB;
    int tid = threadIdx.x;

    if (tid < 16) {
        int g = m0 + tid;
        int valid = (g < count) ? 1 : 0;
        int pos = valid ? list[g] : 0;
        s_pos[tid]   = pos;
        s_row[tid]   = valid ? (tokens[pos] - low) : 0;
        s_valid[tid] = valid;
    }
    __syncthreads();

    const int aM = tid >> 3, aK = (tid & 7) << 2;
    const float* aPtr = emb + (long)s_row[aM] * HSZ + aK;

    auto loadA = [&](int k0) -> f4 {
        f4 r = {0.f, 0.f, 0.f, 0.f};
        if (tid < 128 && (k0 + aK + 4 <= HSZ)) r = *(const f4*)(aPtr + k0);
        return r;
    };
    auto loadB = [&](int k0, int i) -> f4 {
        int s = tid + i * 256;
        f4 r = {0.f, 0.f, 0.f, 0.f};
        if (HEAD) {
            int n = s >> 3, kq = (s & 7) << 2;
            int kk = k0 + kq;
            if (kk + 4 <= HSZ) r = *(const f4*)(w + (long)(n0 + n) * HSZ + kk);
        } else {
            int k = s >> 5, nq = (s & 31) << 2;
            int kk = k0 + k;
            if (kk < HSZ) r = *(const f4*)(w + (long)kk * D_OUT + (n0 + nq));
        }
        return r;
    };

    f4 regA = loadA(0);
    f4 regB[4];
    #pragma unroll
    for (int i = 0; i < 4; ++i) regB[i] = loadB(0, i);

    const int lane = tid & 31, wave = tid >> 5;
    const int mn = lane & 15, hi = (lane >> 4) & 1;
    v8f acc = {};

    for (int kt = 0; kt < KITERS; ++kt) {
        if (tid < 128) {
            u32x2 t; t.x = pack2_bf16(regA.x, regA.y); t.y = pack2_bf16(regA.z, regA.w);
            *(u32x2*)&ldsA[aM * KC + aK] = t;
        }
        #pragma unroll
        for (int i = 0; i < 4; ++i) {
            int s = tid + i * 256;
            if (HEAD) {
                int n = s >> 3, kq = (s & 7) << 2;
                u32x2 t; t.x = pack2_bf16(regB[i].x, regB[i].y); t.y = pack2_bf16(regB[i].z, regB[i].w);
                *(u32x2*)&ldsB[n * KC + kq] = t;
            } else {
                int k = s >> 5, nq = (s & 31) << 2;
                ldsB[(nq + 0) * KC + k] = cvt_bf16(regB[i].x);
                ldsB[(nq + 1) * KC + k] = cvt_bf16(regB[i].y);
                ldsB[(nq + 2) * KC + k] = cvt_bf16(regB[i].z);
                ldsB[(nq + 3) * KC + k] = cvt_bf16(regB[i].w);
            }
        }
        __syncthreads();
        if (kt + 1 < KITERS) {
            int k0 = (kt + 1) * KC;
            regA = loadA(k0);
            #pragma unroll
            for (int i = 0; i < 4; ++i) regB[i] = loadB(k0, i);
        }
        union { unsigned int u[8]; v16bf v; } fa, fb;
        const unsigned int* pa = (const unsigned int*)(ldsA + mn * KC);
        const unsigned int* pb = (const unsigned int*)(ldsB + (wave * 16 + mn) * KC);
        #pragma unroll
        for (int p = 0; p < 8; ++p) {
            int kb = 2 * p + ((p >= 4) ? 8 : 0) + hi * 8;
            fa.u[p] = pa[kb >> 1];
            fb.u[p] = pb[kb >> 1];
        }
        acc = __builtin_amdgcn_wmma_f32_16x16x32_bf16(false, fa.v, false, fb.v,
                                                      (short)0, acc, false, false);
        __syncthreads();
    }

    #pragma unroll
    for (int r = 0; r < 8; ++r)
        ldsC[(r + hi * 8) * NB + wave * 16 + mn] = acc[r];
    __syncthreads();

    #pragma unroll
    for (int i = 0; i < 2; ++i) {
        int e = tid + i * 256;
        int m = e >> 5, cq = (e & 31) << 2;
        if (s_valid[m]) {
            f4 v = *(const f4*)&ldsC[m * NB + cq];
            __builtin_nontemporal_store(v, (f4*)(out + (long)s_pos[m] * D_OUT + n0 + cq));
        }
    }
}

extern "C" void kernel_launch(void* const* d_in, const int* in_sizes, int n_in,
                              void* d_out, int out_size, void* d_ws, size_t ws_size,
                              hipStream_t stream) {
    const int*   tokens    = (const int*)  d_in[0];
    const float* head_emb  = (const float*)d_in[1];
    const float* head_w    = (const float*)d_in[2];
    const float* tail_emb0 = (const float*)d_in[3];
    const float* tail_lin0 = (const float*)d_in[4];
    const float* tail_emb1 = (const float*)d_in[5];
    const float* tail_lin1 = (const float*)d_in[6];
    const float* tail_emb2 = (const float*)d_in[7];
    const float* tail_lin2 = (const float*)d_in[8];
    float* out = (float*)d_out;

    int n = in_sizes[0];                          // 32768 tokens
    int* counters = (int*)d_ws;
    int* lists    = (int*)((char*)d_ws + 64);

    int cap = n;
    {
        size_t need = 64 + (size_t)4 * (size_t)n * sizeof(int);
        if (ws_size < need) {
            size_t avail = (ws_size > 64) ? (ws_size - 64) : 0;
            cap = (int)(avail / (4 * sizeof(int)));
            if (cap < 1) cap = 1;
        }
    }

    size_t lists_end = 64 + (size_t)4 * (size_t)cap * sizeof(int);
    size_t woff = (lists_end + 63) & ~(size_t)63;
    const size_t WE_H = 1024u * 1024u, WE_0 = 1024u * 256u, WE_1 = 1024u * 64u, WE_2 = 1024u * 16u;
    size_t welems = WE_H + WE_0 + WE_1 + WE_2;
    bool pre = (cap == n) && (ws_size >= woff + welems * 2);

    ai_init_counters<<<1, 32, 0, stream>>>(counters);
    ai_partition<<<(n + 255) / 256, 256, 0, stream>>>(tokens, n, counters, lists, cap);

    dim3 grid((n + 15) / 16, D_OUT / 128);

    if (pre) {
        unsigned short* wb  = (unsigned short*)((char*)d_ws + woff);
        unsigned short* wb0 = wb  + WE_H;
        unsigned short* wb1 = wb0 + WE_0;
        unsigned short* wb2 = wb1 + WE_1;
        ai_conv_straight <<<(int)(WE_H / 2048), 256, 0, stream>>>(head_w,    wb,  (int)WE_H);
        ai_conv_transpose<<<(int)(WE_0 / 2048), 256, 0, stream>>>(tail_lin0, wb0, 256);
        ai_conv_transpose<<<(int)(WE_1 / 2048), 256, 0, stream>>>(tail_lin1, wb1, 64);
        ai_conv_transpose<<<(int)(WE_2 / 2048), 256, 0, stream>>>(tail_lin2, wb2, 16);

        ai_cluster_gemm_pre<1024><<<grid, 256, 0, stream>>>(tokens, head_emb,  wb,  counters, 0, lists + 0L * cap, cap, 0,    out);
        ai_cluster_gemm_pre< 256><<<grid, 256, 0, stream>>>(tokens, tail_emb0, wb0, counters, 1, lists + 1L * cap, cap, CUT1, out);
        ai_cluster_gemm_pre<  64><<<grid, 256, 0, stream>>>(tokens, tail_emb1, wb1, counters, 2, lists + 2L * cap, cap, CUT2, out);
        ai_cluster_gemm_pre<  16><<<grid, 256, 0, stream>>>(tokens, tail_emb2, wb2, counters, 3, lists + 3L * cap, cap, CUT3, out);
    } else {
        ai_cluster_gemm_cvt<1024, true ><<<grid, 256, 0, stream>>>(tokens, head_emb,  head_w,    counters, 0, lists + 0L * cap, cap, 0,    out);
        ai_cluster_gemm_cvt< 256, false><<<grid, 256, 0, stream>>>(tokens, tail_emb0, tail_lin0, counters, 1, lists + 1L * cap, cap, CUT1, out);
        ai_cluster_gemm_cvt<  64, false><<<grid, 256, 0, stream>>>(tokens, tail_emb1, tail_lin1, counters, 2, lists + 2L * cap, cap, CUT2, out);
        ai_cluster_gemm_cvt<  16, false><<<grid, 256, 0, stream>>>(tokens, tail_emb2, tail_lin2, counters, 3, lists + 3L * cap, cap, CUT3, out);
    }
}
